// Attention_76476187672744
// MI455X (gfx1250) — compile-verified, matrix operand order checked by reading, not run on hardware
//
#include <hip/hip_runtime.h>

// Problem constants (match reference)
#define BB 4
#define SS 2048
#define DD 1024
#define BSROWS (BB * SS)   // 8192

typedef __attribute__((ext_vector_type(16))) _Float16 v16h;
typedef __attribute__((ext_vector_type(8)))  _Float16 v8h;
typedef __attribute__((ext_vector_type(4)))  _Float16 v4h;
typedef __attribute__((ext_vector_type(8)))  float    v8f;

union FragH { v16h v; uint4 q[2]; };

// ---------------------------------------------------------------------------
// f32 -> f16 conversion (vectorized by 4)
// ---------------------------------------------------------------------------
__global__ void f32_to_f16_kernel(const float* __restrict__ src,
                                  _Float16* __restrict__ dst, int n4) {
  int i = blockIdx.x * blockDim.x + threadIdx.x;
  if (i < n4) {
    float4 f = ((const float4*)src)[i];
    v4h h = { (_Float16)f.x, (_Float16)f.y, (_Float16)f.z, (_Float16)f.w };
    ((v4h*)dst)[i] = h;
  }
}

// ---------------------------------------------------------------------------
// Generic WMMA GEMM:  C[M,N] = scale * (A[M,K] * Bt[N,K]^T) + bias[N]
//   A, Bt: f16 row-major (Bt row n holds B column n -> matches torch Linear W)
//   out_mode 0: f32 row-major   (ldc = row stride)
//   out_mode 1: f16 row-major
//   out_mode 2: f16 transposed  (C^T[N,M], ldc = M-extent stride)
// Block tile 128x128, BK=32, 256 threads = 8 waves (4 in M x 2 in N),
// each wave: 2x4 tiles of v_wmma_f32_16x16x32_f16.
// Tile fill uses GLOBAL_LOAD_ASYNC_TO_LDS_B128 (ASYNCcnt) with LDS double
// buffering; last K-step is peeled so the hot loop is branch-free.
// M, N, K must be multiples of 128/128/32 (true for all uses here).
// ---------------------------------------------------------------------------
#define BM 128
#define BN 128
#define BK 32
#define LDSS 40               // LDS row stride in halves: 80B, conflict-free
#define TILEH (BM * LDSS)     // halves per tile (A or B)
#define BUFH  (2 * TILEH)     // halves per double-buffer stage (A + B)

__launch_bounds__(256)
__global__ void wmma_gemm_kernel(const _Float16* __restrict__ A, int lda,
                                 const _Float16* __restrict__ Bt, int ldb,
                                 const float* __restrict__ bias,
                                 float scale,
                                 void* __restrict__ Cout, int ldc,
                                 int Kdim, int out_mode) {
  __shared__ _Float16 smem[2 * BUFH];   // [buf][A tile | B tile]

  const int tid  = threadIdx.x;
  const int lane = tid & 31;
  const int wave = tid >> 5;
  const int wm   = wave >> 1;     // 0..3 : wave row in block tile
  const int wn   = wave & 1;      // 0..1 : wave col in block tile
  const int nl   = lane & 15;     // N (or M-for-A) within 16x16 tile
  const int hi   = lane >> 4;     // lane-group select (K/M halves)

  const int m0 = blockIdx.x * BM;
  const int n0 = blockIdx.y * BN;

  // cooperative tile-fill mapping: 2 threads per row, 32 B (2x b128) each
  const int ldrow = tid >> 1;          // 0..127
  const int ldseg = (tid & 1) * 16;    // 0 or 16 halves

  const _Float16* gA = A  + (size_t)(m0 + ldrow) * lda + ldseg;
  const _Float16* gB = Bt + (size_t)(n0 + ldrow) * ldb + ldseg;

  // 32-bit LDS byte addresses for this thread's fill slots
  const unsigned ldsBase = (unsigned)(unsigned long long)(&smem[0]);
  const unsigned slot    = (unsigned)(ldrow * LDSS + ldseg) * 2u;
  const unsigned sA0     = ldsBase + slot;                 // buf0 A
  const unsigned sB0     = ldsBase + (unsigned)TILEH * 2u + slot;

  // issue this thread's 4 async b128 copies for buffer `buf`, k-offset `kb`
  auto issue_tile = [&](int buf, int kb) {
    unsigned long long pa = (unsigned long long)(gA + kb);
    unsigned long long pb = (unsigned long long)(gB + kb);
    unsigned sa = sA0 + (unsigned)buf * (unsigned)BUFH * 2u;
    unsigned sb = sB0 + (unsigned)buf * (unsigned)BUFH * 2u;
    // offset: immediate applies to BOTH the LDS and global address (ISA 10.x)
    asm volatile("global_load_async_to_lds_b128 %0, %1, off"
                 :: "v"(sa), "v"(pa) : "memory");
    asm volatile("global_load_async_to_lds_b128 %0, %1, off offset:16"
                 :: "v"(sa), "v"(pa) : "memory");
    asm volatile("global_load_async_to_lds_b128 %0, %1, off"
                 :: "v"(sb), "v"(pb) : "memory");
    asm volatile("global_load_async_to_lds_b128 %0, %1, off offset:16"
                 :: "v"(sb), "v"(pb) : "memory");
  };

  v8f acc[2][4] = {};

  // consume buffer `buf`: 12x ds_load_b128 fragment fetch + 8x WMMA
  auto compute = [&](int buf) {
    const _Float16* Abuf = smem + buf * BUFH;
    const _Float16* Bbuf = Abuf + TILEH;
    FragH af[2], bf[4];
#pragma unroll
    for (int tm = 0; tm < 2; ++tm) {
      const _Float16* p = &Abuf[(wm * 32 + tm * 16 + nl) * LDSS + hi * 8];
      af[tm].q[0] = *(const uint4*)p;          // K = hi*8 .. hi*8+7
      af[tm].q[1] = *(const uint4*)(p + 16);   // K = 16+hi*8 .. 16+hi*8+7
    }
#pragma unroll
    for (int tn = 0; tn < 4; ++tn) {
      const _Float16* p = &Bbuf[(wn * 64 + tn * 16 + nl) * LDSS + hi * 8];
      bf[tn].q[0] = *(const uint4*)p;
      bf[tn].q[1] = *(const uint4*)(p + 16);
    }
#pragma unroll
    for (int tm = 0; tm < 2; ++tm)
#pragma unroll
      for (int tn = 0; tn < 4; ++tn)
        acc[tm][tn] = __builtin_amdgcn_wmma_f32_16x16x32_f16(
            false, af[tm].v, false, bf[tn].v, (short)0, acc[tm][tn],
            false, false);
  };

  issue_tile(0, 0);   // prefetch first tile into buffer 0

  // Branch-free steady state: all iterations except the last K-step.
  int ib = 0;
  int kb = 0;
  for (; kb + BK < Kdim; kb += BK, ib ^= 1) {
    // (1) all waves done READING buffer ib^1 (consumed last iter);
    //     __syncthreads lowers with s_wait_dscnt 0 so frag loads completed.
    __syncthreads();
    issue_tile(ib ^ 1, kb + BK);
    // 4 newest outstanding = next buffer; async loads complete in order,
    // so <=4 means this buffer's 4 copies have landed in LDS.
    asm volatile("s_wait_asynccnt 0x4" ::: "memory");
    // (2) everyone's copies for buffer ib are visible
    __syncthreads();
    compute(ib);
  }

  // Peeled final K-step: drain ASYNCcnt fully, no further issues.
  asm volatile("s_wait_asynccnt 0x0" ::: "memory");
  __syncthreads();
  compute(ib);

  // Epilogue. C fragment: lane -> N = nl, element r -> M = r + 8*hi.
  const bool has_bias = (bias != nullptr);
#pragma unroll
  for (int tm = 0; tm < 2; ++tm) {
#pragma unroll
    for (int tn = 0; tn < 4; ++tn) {
      const int mb = m0 + wm * 32 + tm * 16 + hi * 8;
      const int nb = n0 + wn * 64 + tn * 16 + nl;
      const float bv = has_bias ? bias[nb] : 0.0f;
      v8f c = acc[tm][tn];
      if (out_mode == 0) {
        float* C = (float*)Cout;
#pragma unroll
        for (int r = 0; r < 8; ++r)
          C[(size_t)(mb + r) * ldc + nb] = c[r] * scale + bv;
      } else if (out_mode == 1) {
        _Float16* C = (_Float16*)Cout;
#pragma unroll
        for (int r = 0; r < 8; ++r)
          C[(size_t)(mb + r) * ldc + nb] = (_Float16)(c[r] * scale + bv);
      } else {
        _Float16* C = (_Float16*)Cout;          // transposed store: row nb
        v8h t;
#pragma unroll
        for (int r = 0; r < 8; ++r) t[r] = (_Float16)(c[r] * scale + bv);
        *(v8h*)(&C[(size_t)nb * ldc + mb]) = t; // 8 contiguous M values, 16B
      }
    }
  }
}

// ---------------------------------------------------------------------------
// Row softmax over SS=2048 columns: f32 logits -> f16 probs. Block per row.
// ---------------------------------------------------------------------------
__launch_bounds__(256)
__global__ void softmax_kernel(const float* __restrict__ logits,
                               _Float16* __restrict__ probs) {
  __shared__ float red[256];
  const int row = blockIdx.x;
  const int tid = threadIdx.x;
  const float* rp = logits + (size_t)row * SS;

  float vals[SS / 256];
  float m = -3.0e38f;
#pragma unroll
  for (int i = 0; i < SS / 256; ++i) {
    vals[i] = rp[tid + i * 256];
    m = fmaxf(m, vals[i]);
  }
  red[tid] = m;
  __syncthreads();
  for (int s = 128; s > 0; s >>= 1) {
    if (tid < s) red[tid] = fmaxf(red[tid], red[tid + s]);
    __syncthreads();
  }
  m = red[0];
  __syncthreads();

  float sum = 0.0f;
#pragma unroll
  for (int i = 0; i < SS / 256; ++i) {
    vals[i] = __expf(vals[i] - m);
    sum += vals[i];
  }
  red[tid] = sum;
  __syncthreads();
  for (int s = 128; s > 0; s >>= 1) {
    if (tid < s) red[tid] += red[tid + s];
    __syncthreads();
  }
  const float inv = 1.0f / red[0];

  _Float16* op = probs + (size_t)row * SS;
#pragma unroll
  for (int i = 0; i < SS / 256; ++i)
    op[tid + i * 256] = (_Float16)(vals[i] * inv);
}

// ---------------------------------------------------------------------------
// Launcher. Workspace layout (f16 unless noted), ~94 MB total:
//   xh[8192*1024] | wqh,wkh,wvh[1024*1024 each] | qh,kh[8192*1024 each]
//   | vth[4*1024*2048] (V^T per batch) | logits f32[2048*2048] | ph[2048*2048]
// ---------------------------------------------------------------------------
extern "C" void kernel_launch(void* const* d_in, const int* in_sizes, int n_in,
                              void* d_out, int out_size, void* d_ws,
                              size_t ws_size, hipStream_t stream) {
  const float* x  = (const float*)d_in[0];
  const float* Wq = (const float*)d_in[1];
  const float* bq = (const float*)d_in[2];
  const float* Wk = (const float*)d_in[3];
  const float* bk = (const float*)d_in[4];
  const float* Wv = (const float*)d_in[5];
  const float* bv = (const float*)d_in[6];
  float* out = (float*)d_out;

  _Float16* xh  = (_Float16*)d_ws;
  _Float16* wqh = xh  + (size_t)BSROWS * DD;
  _Float16* wkh = wqh + (size_t)DD * DD;
  _Float16* wvh = wkh + (size_t)DD * DD;
  _Float16* qh  = wvh + (size_t)DD * DD;
  _Float16* kh  = qh  + (size_t)BSROWS * DD;
  _Float16* vth = kh  + (size_t)BSROWS * DD;               // [B][D][S]
  float*    logits = (float*)(vth + (size_t)BB * DD * SS); // [S][S] reused
  _Float16* ph  = (_Float16*)(logits + (size_t)SS * SS);   // [S][S] reused

  // 1) convert x, Wq, Wk, Wv to f16
  {
    int n4 = BSROWS * DD / 4;
    f32_to_f16_kernel<<<(n4 + 255) / 256, 256, 0, stream>>>(x, xh, n4);
    int w4 = DD * DD / 4;
    f32_to_f16_kernel<<<(w4 + 255) / 256, 256, 0, stream>>>(Wq, wqh, w4);
    f32_to_f16_kernel<<<(w4 + 255) / 256, 256, 0, stream>>>(Wk, wkh, w4);
    f32_to_f16_kernel<<<(w4 + 255) / 256, 256, 0, stream>>>(Wv, wvh, w4);
  }

  const dim3 blk(256);
  const float rs = 0.03125f;  // 1/sqrt(1024)

  // 2) Q = x*Wq^T+bq, K = x*Wk^T+bk  (f16 row-major)
  wmma_gemm_kernel<<<dim3(BSROWS / BM, DD / BN), blk, 0, stream>>>(
      xh, DD, wqh, DD, bq, 1.0f, qh, DD, DD, 1);
  wmma_gemm_kernel<<<dim3(BSROWS / BM, DD / BN), blk, 0, stream>>>(
      xh, DD, wkh, DD, bk, 1.0f, kh, DD, DD, 1);

  // 3) V per batch, stored transposed: vth[b][d][s]
  for (int b = 0; b < BB; ++b)
    wmma_gemm_kernel<<<dim3(SS / BM, DD / BN), blk, 0, stream>>>(
        xh + (size_t)b * SS * DD, DD, wvh, DD, bv, 1.0f,
        vth + (size_t)b * DD * SS, SS, DD, 2);

  // 4) per batch: logits = Q K^T / 32 -> softmax -> out = P V
  for (int b = 0; b < BB; ++b) {
    wmma_gemm_kernel<<<dim3(SS / BM, SS / BN), blk, 0, stream>>>(
        qh + (size_t)b * SS * DD, DD, kh + (size_t)b * SS * DD, DD,
        nullptr, rs, logits, SS, DD, 0);
    softmax_kernel<<<SS, 256, 0, stream>>>(logits, ph);
    wmma_gemm_kernel<<<dim3(SS / BM, DD / BN), blk, 0, stream>>>(
        ph, SS, vth + (size_t)b * DD * SS, SS,
        nullptr, 1.0f, out + (size_t)b * SS * DD, DD, SS, 0);
  }
}
